// OrthogonalSylvesterVAE_86268713108076
// MI455X (gfx1250) — compile-verified, matrix operand order checked by reading, not run on hardware
//
#include <hip/hip_runtime.h>
#include <string.h>

// ---------------- CDNA5 wave32 WMMA types ----------------
typedef __attribute__((ext_vector_type(16))) _Float16 v16h;
typedef __attribute__((ext_vector_type(8)))  _Float16 v8h;
typedef __attribute__((ext_vector_type(8)))  float    v8f;
union F16x16 { v16h v; v8h h[2]; };

#define EPS_BN 1e-5f

// ---------------- workspace layout (bytes) ----------------
static const size_t W_C1  = 0;         // 32x32  f16
static const size_t W_C2  = 4096;      // 160x16 f16
static const size_t W_C3  = 12288;     // 128x16 f16
static const size_t W_D1  = 16384;     // 64x16  f16
static const size_t W_D2  = 20480;     // 96x32  f16
static const size_t W_D3  = 28672;     // 96x16  f16
static const size_t W_P1  = 32768;     // 32x128 f16
static const size_t W_P2  = 40960;     // 128x1152 f16 (294912)
static const size_t W_AM  = 335872;    // 1120x512 f16 (1146880)
static const size_t B_AM  = 1482752;   // 512 f32
static const size_t B_P1  = 1484800;   // 128 f32
static const size_t B_P2  = 1485312;   // 1152 f32
static const size_t STB   = 1489920;   // stats region (f32[5536])
// stats float offsets inside STB
#define S1SUM 0
#define S1SQ  32
#define S1SC  64
#define S1SH  96
#define S2SUM 128
#define S2SQ  144
#define S2SC  160
#define S2SH  176
#define S3SUM 192
#define S3SQ  200
#define S3SC  208
#define S3SH  216
#define P1SUM 224
#define P1SQ  352
#define P1SC  480
#define P1SH  608
#define P2SUM 736
#define P2SQ  1888
#define P2A2  3040
#define P2C2  4192
#define D1SUM 5344
#define D1SQ  5360
#define D1SC  5376
#define D1SH  5392
#define D2SUM 5408
#define D2SQ  5440
#define D2SC  5472
#define D2SH  5504
#define NSTATS 5536
// activation buffers
static const size_t BUF1 = 2097152;                         // y1/a1 (f16), later e2 (f16)
static const size_t BUF2 = BUF1 + 77594624;                 // y2/a2 (f16), later d2raw(f32), later e1(f16)
static const size_t BUF3 = BUF2 + 37748736;                 // hA (f16, lda=1120), later d2h (f16)
static const size_t BUF4 = BUF3 + 18350080;                 // 24 MB region
static const size_t OF_AMORC = BUF4;                        // 8192x512 f32
static const size_t OF_ZH    = BUF4 + 17000448;             // 8192x32 f16
static const size_t OF_D1RAW = BUF4 + 17524736;             // 8192x128 f32
static const size_t OF_D1H   = BUF4 + 21719040;             // 8192x128 f16

// output (float) offsets
static const size_t O_MU  = 4915200;
static const size_t O_LV  = 5038080;
static const size_t O_LDJ = 5160960;
static const size_t O_Z0  = 5169152;
static const size_t O_Z   = 5292032;

// ---------------- parameter bundle (order == setup_inputs order) ----------------
struct P {
  const float *x, *eps;
  const float *conv1_w, *conv1_b, *bn1_g, *bn1_b;
  const float *conv2_w, *conv2_b, *bn2_g, *bn2_b;
  const float *conv3_w, *conv3_b, *bn3_g, *bn3_b;
  const float *qz_mean_w, *qz_mean_b, *qz_logvar_w, *qz_logvar_b;
  const float *amor_d_w, *amor_d_b, *amor_diag1_w, *amor_diag1_b;
  const float *amor_diag2_w, *amor_diag2_b, *amor_q_w, *amor_q_b;
  const float *amor_b_w, *amor_b_b;
  const float *pxl1_w, *pxl1_b, *pxbn1_g, *pxbn1_b;
  const float *pxl2_w, *pxl2_b, *pxbn2_g, *pxbn2_b;
  const float *pxbn0_g, *pxbn0_b;
  const float *dconv1_w, *dconv1_b, *dbn1_g, *dbn1_b;
  const float *dconv2_w, *dconv2_b, *dbn2_g, *dbn2_b;
  const float *dconv3_w, *dconv3_b;
};
static_assert(sizeof(P) == 48 * sizeof(void*), "P layout");

// ---------------- prep: f16 effective weights + fused matrices + zero stats ----------------
__global__ __launch_bounds__(256) void prep_kernel(P p, char* ws) {
  const long gs  = (long)gridDim.x * 256;
  const long tid = (long)blockIdx.x * 256 + threadIdx.x;

  _Float16* c1 = (_Float16*)(ws + W_C1);
  for (long i = tid; i < 32 * 32; i += gs) {
    int k = (int)(i / 32), oc = (int)(i % 32);
    c1[i] = (_Float16)((k < 12) ? p.conv1_w[oc * 12 + k] : 0.f);
  }
  _Float16* c2 = (_Float16*)(ws + W_C2);
  for (long i = tid; i < 160 * 16; i += gs) {
    int k = (int)(i / 16), oc = (int)(i % 16);
    c2[i] = (_Float16)(p.conv2_w[oc * 160 + k]);
  }
  _Float16* c3 = (_Float16*)(ws + W_C3);
  for (long i = tid; i < 128 * 16; i += gs) {
    int k = (int)(i / 16), oc = (int)(i % 16);
    c3[i] = (_Float16)((k < 112 && oc < 8) ? p.conv3_w[oc * 112 + k] : 0.f);
  }
  _Float16* dw1 = (_Float16*)(ws + W_D1);
  for (long i = tid; i < 64 * 16; i += gs) {
    int k = (int)(i / 16), oc = (int)(i % 16);
    float v = 0.f;
    if (k < 56) { int ic = k / 7, j = k % 7; v = p.dconv1_w[ic * 112 + oc * 7 + (6 - j)]; }
    dw1[i] = (_Float16)v;
  }
  _Float16* dw2 = (_Float16*)(ws + W_D2);
  for (long i = tid; i < 96 * 32; i += gs) {
    int k = (int)(i / 32), oc = (int)(i % 32);
    float v = 0.f;
    if (k < 80) { int ic = k / 5, j = k % 5; v = p.dconv2_w[ic * 160 + oc * 5 + (4 - j)]; }
    dw2[i] = (_Float16)v;
  }
  _Float16* dw3 = (_Float16*)(ws + W_D3);
  for (long i = tid; i < 96 * 16; i += gs) {
    int k = (int)(i / 16), n = (int)(i % 16);
    float v = 0.f;
    if (n < 4) { int ic = k / 3, j = k % 3; v = p.dconv3_w[ic * 12 + (2 - j) * 4 + n]; }
    dw3[i] = (_Float16)v;
  }
  _Float16* p1 = (_Float16*)(ws + W_P1);
  for (long i = tid; i < 32 * 128; i += gs) {
    int k = (int)(i / 128), n = (int)(i % 128);
    p1[i] = (_Float16)((k < 15 && n < 100) ? p.pxl1_w[k * 100 + n] : 0.f);
  }
  _Float16* p2 = (_Float16*)(ws + W_P2);
  for (long i = tid; i < 128 * 1152; i += gs) {
    int k = (int)(i / 1152), n = (int)(i % 1152);
    p2[i] = (_Float16)((k < 100 && n < 1104) ? p.pxl2_w[k * 1104 + n] : 0.f);
  }
  _Float16* am = (_Float16*)(ws + W_AM);
  for (long i = tid; i < 1120L * 512; i += gs) {
    int k = (int)(i / 512), n = (int)(i % 512);
    float v = 0.f;
    if (k < 1104) {
      if      (n < 15)  v = p.qz_mean_w[k * 15 + n];
      else if (n < 30)  v = p.qz_logvar_w[k * 15 + (n - 15)];
      else if (n < 130) v = p.amor_d_w[k * 100 + (n - 30)];
      else if (n < 150) v = p.amor_diag1_w[k * 20 + (n - 130)];
      else if (n < 170) v = p.amor_diag2_w[k * 20 + (n - 150)];
      else if (n < 470) v = p.amor_q_w[k * 300 + (n - 170)];
      else if (n < 490) v = p.amor_b_w[k * 20 + (n - 470)];
    }
    am[i] = (_Float16)v;
  }
  float* amb = (float*)(ws + B_AM);
  for (long i = tid; i < 512; i += gs) {
    int n = (int)i; float v = 0.f;
    if      (n < 15)  v = p.qz_mean_b[n];
    else if (n < 30)  v = p.qz_logvar_b[n - 15];
    else if (n < 130) v = p.amor_d_b[n - 30];
    else if (n < 150) v = p.amor_diag1_b[n - 130];
    else if (n < 170) v = p.amor_diag2_b[n - 150];
    else if (n < 470) v = p.amor_q_b[n - 170];
    else if (n < 490) v = p.amor_b_b[n - 470];
    amb[n] = v;
  }
  float* p1b = (float*)(ws + B_P1);
  for (long i = tid; i < 128; i += gs) p1b[i] = (i < 100) ? p.pxl1_b[i] : 0.f;
  float* p2b = (float*)(ws + B_P2);
  for (long i = tid; i < 1152; i += gs) p2b[i] = (i < 1104) ? p.pxl2_b[i] : 0.f;

  float* st = (float*)(ws + STB);
  for (long i = tid; i < NSTATS; i += gs) st[i] = 0.f;

  // zero the K padding columns (1104..1119) of hA once per call
  _Float16* hA = (_Float16*)(ws + BUF3);
  for (long i = tid; i < 8192L * 16; i += gs)
    hA[(i >> 4) * 1120 + 1104 + (i & 15)] = (_Float16)0.f;
}

// ---------------- dense WMMA GEMM (async-LDS double-buffered) ----------------
// C(MxN,f32) = A(MxK,f16) * B(KxN,f16) + bias ; A tile staged with
// global_load_async_to_lds_b128 (ASYNCcnt) and ping-pong LDS buffers so the
// next K-tile's copy overlaps the current tile's WMMAs.
template<int NT, bool STATS>
__global__ __launch_bounds__(256) void dense_gemm(
    const _Float16* __restrict__ A, int lda,
    const _Float16* __restrict__ Bw, int ldb,
    const float* __restrict__ bias,
    float* __restrict__ C, int ldc, int K,
    float* __restrict__ ssum, float* __restrict__ ssq, int nreal)
{
  __shared__ _Float16 sA[2][128 * 40];
  __shared__ _Float16 sB[2][NT * 16 * 40];
  const int  tid = threadIdx.x;
  const long m0  = (long)blockIdx.x * 128;
  const int  n0  = blockIdx.y * (NT * 16);

  v8f acc[NT] = {};
  const int r = tid >> 1, part = tid & 1;

  auto issueA = [&](int k0, int buf) {
    unsigned long long g0 =
        (unsigned long long)(A + (size_t)(m0 + r) * lda + k0 + part * 16);
    unsigned lds0 = (unsigned)(size_t)&sA[buf][r * 40 + part * 16];
    asm volatile("global_load_async_to_lds_b128 %0, %1, off"
                 :: "v"(lds0), "v"(g0) : "memory");
    asm volatile("global_load_async_to_lds_b128 %0, %1, off offset:16"
                 :: "v"(lds0), "v"(g0) : "memory");
  };
  auto loadB = [&](int k0, int buf) {
    const int nb = NT * 16 * 32 / 8;
    if (tid < nb) {
      int k = tid & 31, nc = (tid >> 5) * 8;
      float4 v = *(const float4*)(Bw + (size_t)(k0 + k) * ldb + n0 + nc);
      const _Float16* hv = (const _Float16*)&v;
      #pragma unroll
      for (int j = 0; j < 8; j++) sB[buf][(nc + j) * 40 + k] = hv[j];
    }
  };

  issueA(0, 0);
  loadB(0, 0);
  const int nsteps = K >> 5;
  for (int s = 0; s < nsteps; s++) {
    const int k0 = s << 5, cur = s & 1;
    asm volatile("s_wait_asynccnt 0x0" ::: "memory");
    __syncthreads();
    if (s + 1 < nsteps) { issueA(k0 + 32, cur ^ 1); loadB(k0 + 32, cur ^ 1); }

    const int wv = tid >> 5, ln = tid & 31;
    const int lrow = ln & 15, hi = ln >> 4;
    F16x16 afr;
    const _Float16* ab = &sA[cur][(wv * 16 + lrow) * 40 + hi * 8];
    afr.h[0] = *(const v8h*)(ab);
    afr.h[1] = *(const v8h*)(ab + 16);
    #pragma unroll
    for (int nt = 0; nt < NT; nt++) {
      F16x16 bfr;
      const _Float16* bb = &sB[cur][(nt * 16 + lrow) * 40 + hi * 8];
      bfr.h[0] = *(const v8h*)(bb);
      bfr.h[1] = *(const v8h*)(bb + 16);
      acc[nt] = __builtin_amdgcn_wmma_f32_16x16x32_f16(
          false, afr.v, false, bfr.v, (short)0, acc[nt], false, false);
    }
    __syncthreads();
  }
  // epilogue: lane holds column (lane&15), rows e + (lane>>4)*8
  const int wv = tid >> 5, ln = tid & 31;
  const int col = ln & 15, rhi = (ln >> 4) * 8;
  #pragma unroll
  for (int nt = 0; nt < NT; nt++) {
    const int n = n0 + nt * 16 + col;
    float psum = 0.f, psq = 0.f;
    const float bv = bias[n];
    const int mb = (int)m0 + wv * 16 + rhi;
    #pragma unroll
    for (int e = 0; e < 8; e++) {
      float v = acc[nt][e] + bv;
      C[(size_t)(mb + e) * ldc + n] = v;
      psum += v; psq += v * v;
    }
    if (STATS && n < nreal) { atomicAdd(&ssum[n], psum); atomicAdd(&ssq[n], psq); }
  }
}

// ---------------- implicit-GEMM conv / convT (WMMA) ----------------
// M = 8192*LOUT rows (b,t); N = NT*16 output cols; K = KTOT (padded to KP).
template<int KTOT, int IC, int KL, int PAD, int LIN, int LOUT, int NT, int OCR, bool SRCX, bool XDEC>
__global__ __launch_bounds__(256) void conv_gemm(
    const void* __restrict__ srcv, const _Float16* __restrict__ weff,
    _Float16* __restrict__ dsth, int bstride,
    float* __restrict__ dstf, const float* __restrict__ biasp,
    float* __restrict__ ssum, float* __restrict__ ssq)
{
  constexpr int KP = (KTOT + 31) / 32 * 32;
  constexpr int KS = KP + 8;
  __shared__ _Float16 sA[128 * 40];
  __shared__ _Float16 sB[NT * 16 * KS];
  const int tid = threadIdx.x;
  const int m0  = (int)blockIdx.x * 128;
  const float* srcx = (const float*)srcv;
  const _Float16* srch = (const _Float16*)srcv;

  // weights -> LDS, transposed [col][k]
  for (int i = tid; i < NT * 16 * KP; i += 256) {
    int col = i / KP, k = i % KP;
    sB[col * KS + k] = weff[k * (NT * 16) + col];
  }
  v8f acc[NT] = {};
  const int r = tid >> 1, part = tid & 1;
  const int mrow = m0 + r;
  const int bb = mrow / LOUT;
  const int tp = mrow - bb * LOUT;
  __syncthreads();

  for (int k0 = 0; k0 < KP; k0 += 32) {
    // gather patch row into sA
    {
      if (!SRCX && k0 + 32 < KP)
        __builtin_prefetch(srch + (size_t)bb * (IC * LIN) + ((k0 + 32) / KL) * LIN + tp, 0, 0);
      _Float16* dst = &sA[r * 40 + part * 16];
      #pragma unroll
      for (int q = 0; q < 16; q++) {
        int k = k0 + part * 16 + q;
        float v = 0.f;
        if (k < KTOT) {
          if (SRCX) {
            int j = k >> 2, w = k & 3;
            int ti = tp + j;
            v = srcx[((size_t)bb * LIN + ti) * 4 + w];
          } else {
            int ic = k / KL, j = k % KL;
            int ti = tp + j - PAD;
            if (ti >= 0 && ti < LIN)
              v = (float)srch[(size_t)bb * (IC * LIN) + ic * LIN + ti];
          }
        }
        dst[q] = (_Float16)v;
      }
    }
    __syncthreads();
    const int wv = tid >> 5, ln = tid & 31;
    const int lrow = ln & 15, hi = ln >> 4;
    F16x16 afr;
    const _Float16* ab = &sA[(wv * 16 + lrow) * 40 + hi * 8];
    afr.h[0] = *(const v8h*)(ab);
    afr.h[1] = *(const v8h*)(ab + 16);
    #pragma unroll
    for (int nt = 0; nt < NT; nt++) {
      F16x16 bfr;
      const _Float16* bp = &sB[(nt * 16 + lrow) * KS + k0 + hi * 8];
      bfr.h[0] = *(const v8h*)(bp);
      bfr.h[1] = *(const v8h*)(bp + 16);
      acc[nt] = __builtin_amdgcn_wmma_f32_16x16x32_f16(
          false, afr.v, false, bfr.v, (short)0, acc[nt], false, false);
    }
    __syncthreads();
  }
  // epilogue: one 32-bit divmod per lane, incremental row->(b,t) after that
  const int wv = tid >> 5, ln = tid & 31;
  const int col = ln & 15, rhi = (ln >> 4) * 8;
  const float b0 = XDEC ? biasp[0] : 0.f;
  const int mb = m0 + wv * 16 + rhi;
  const int b2base = mb / LOUT;
  const int t2base = mb - b2base * LOUT;
  #pragma unroll
  for (int nt = 0; nt < NT; nt++) {
    const int n = nt * 16 + col;
    float psum = 0.f, psq = 0.f;
    int b2 = b2base, t2 = t2base;
    #pragma unroll
    for (int e = 0; e < 8; e++) {
      float v = acc[nt][e];
      if (XDEC) {
        if (n < OCR) dstf[((size_t)b2 * 150 + t2) * 4 + n] = v + b0;
      } else if (n < OCR) {
        dsth[(size_t)b2 * bstride + n * LOUT + t2] = (_Float16)v;
        psum += v; psq += v * v;
      }
      t2++; if (t2 == LOUT) { t2 = 0; b2++; }
    }
    if (!XDEC && n < OCR) { atomicAdd(&ssum[n], psum); atomicAdd(&ssq[n], psq); }
  }
}

// ---------------- BN finalize / apply ----------------
__global__ __launch_bounds__(256) void finalize_bn(
    const float* sum, const float* sq, const float* g, const float* b,
    float* sc, float* sh, int n, float inv)
{
  for (int i = threadIdx.x; i < n; i += 256) {
    float m = sum[i] * inv;
    float var = sq[i] * inv - m * m;
    float s = g[i] * rsqrtf(var + EPS_BN);
    sc[i] = s;
    sh[i] = b[i] - s * m;
  }
}

template<bool LRELU, int C, int L>
__global__ __launch_bounds__(256) void apply_bn_act(
    _Float16* act, const float* sc, const float* sh, int bstride)
{
  const int total = 8192 * C * L;
  const int gs = gridDim.x * 256;
  for (int i = blockIdx.x * 256 + threadIdx.x; i < total; i += gs) {
    int b = i / (C * L);
    int rr = i - b * (C * L);
    int c = rr / L, t = rr - c * L;
    size_t a = (size_t)b * bstride + c * L + t;
    float v = sc[c] * (float)act[a] + sh[c];
    if (LRELU) v = (v > 0.f) ? v : 0.01f * v;
    act[a] = (_Float16)v;
  }
}

__global__ __launch_bounds__(256) void apply_d1(
    const float* __restrict__ d1raw, const float* __restrict__ sc,
    const float* __restrict__ sh, _Float16* __restrict__ d1h)
{
  const int gs = gridDim.x * 256;
  for (int i = blockIdx.x * 256 + threadIdx.x; i < 8192 * 128; i += gs) {
    int n = i & 127;
    float v = 0.f;
    if (n < 100) {
      v = sc[n] * d1raw[i] + sh[n];
      v = (v > 0.f) ? v : 0.01f * v;
    }
    d1h[i] = (_Float16)v;
  }
}

// fused pxbn2 (per-feature) ∘ pxbn0 (per-channel) -> one affine per feature (exact)
__global__ __launch_bounds__(160) void finalize_px(
    const float* sum, const float* sq,
    const float* g2, const float* b2, const float* g0, const float* b0,
    float* A2, float* C2)
{
  int ch = blockIdx.x;           // 0..7
  int f = threadIdx.x;           // 0..159 (138 used)
  __shared__ float sEy[138], sEy2[138];
  __shared__ float mu, rs0;
  float a = 0.f, mf = 0.f;
  if (f < 138) {
    int idx = ch * 138 + f;
    mf = sum[idx] * (1.f / 8192.f);
    float var = sq[idx] * (1.f / 8192.f) - mf * mf;
    float rs = rsqrtf(var + EPS_BN);
    a = g2[idx] * rs;
    sEy[f]  = b2[idx];
    sEy2[f] = b2[idx] * b2[idx] + g2[idx] * g2[idx] * var * rs * rs;
  }
  __syncthreads();
  if (threadIdx.x == 0) {
    float s1 = 0.f, s2 = 0.f;
    for (int i = 0; i < 138; i++) { s1 += sEy[i]; s2 += sEy2[i]; }
    mu = s1 / 138.f;
    float var = s2 / 138.f - mu * mu;
    rs0 = rsqrtf(var + EPS_BN);
  }
  __syncthreads();
  if (f < 138) {
    int idx = ch * 138 + f;
    float A2v = a * g0[ch] * rs0;
    A2[idx] = A2v;
    C2[idx] = g0[ch] * rs0 * (sEy[f] - mu) - A2v * mf + b0[ch];
  }
}

__global__ __launch_bounds__(256) void apply_d2(
    const float* __restrict__ d2raw, const float* __restrict__ A2,
    const float* __restrict__ C2, _Float16* __restrict__ d2h)
{
  const int gs = gridDim.x * 256;
  for (int i = blockIdx.x * 256 + threadIdx.x; i < 8192 * 1104; i += gs) {
    int b = i / 1104; int f = i - b * 1104;
    d2h[i] = (_Float16)(A2[f] * d2raw[(size_t)b * 1152 + f] + C2[f]);
  }
}

// ---------------- orthogonalize + Sylvester flows (one lane per batch element) ----------------
__global__ __launch_bounds__(256) void flows_kernel(
    const float* __restrict__ amorC, const float* __restrict__ eps,
    float* __restrict__ out, _Float16* __restrict__ zh)
{
  int b = blockIdx.x * 256 + threadIdx.x;
  if (b >= 8192) return;
  const float* row = amorC + (size_t)b * 512;
  float zmu[15], zlv[15], z[15];
  #pragma unroll
  for (int i = 0; i < 15; i++) {
    zmu[i] = row[i];
    zlv[i] = row[15 + i];
    z[i] = zmu[i] + eps[b * 15 + i] * expf(0.5f * zlv[i]);
    out[O_MU + (size_t)b * 15 + i] = zmu[i];
    out[O_LV + (size_t)b * 15 + i] = zlv[i];
    out[O_Z0 + (size_t)b * 15 + i] = z[i];
  }
  float ldj = 0.f;
  for (int f = 0; f < 4; f++) {
    float A[75];
    float nrm = 0.f;
    #pragma unroll
    for (int i = 0; i < 75; i++) { float v = row[170 + f * 75 + i]; A[i] = v; nrm += v * v; }
    nrm = rsqrtf(nrm);
    #pragma unroll
    for (int i = 0; i < 75; i++) A[i] *= nrm;
    #pragma clang loop unroll(disable)
    for (int it = 0; it < 100; it++) {
      float T[25];
      #pragma unroll
      for (int m = 0; m < 5; m++)
        #pragma unroll
        for (int n = 0; n < 5; n++) {
          float s = 0.f;
          #pragma unroll
          for (int zz = 0; zz < 15; zz++) s += A[zz * 5 + m] * A[zz * 5 + n];
          T[m * 5 + n] = ((m == n) ? 1.f : 0.f) - s;
        }
      float NA[75];
      #pragma unroll
      for (int zz = 0; zz < 15; zz++)
        #pragma unroll
        for (int m = 0; m < 5; m++) {
          float s = A[zz * 5 + m];
          #pragma unroll
          for (int n = 0; n < 5; n++) s += A[zz * 5 + n] * 0.5f * T[n * 5 + m];
          NA[zz * 5 + m] = s;
        }
      #pragma unroll
      for (int i = 0; i < 75; i++) A[i] = NA[i];
    }
    float d1v[5], d2v[5], bv[5];
    #pragma unroll
    for (int m = 0; m < 5; m++) {
      d1v[m] = tanhf(row[130 + m * 4 + f]);
      d2v[m] = tanhf(row[150 + m * 4 + f]);
      bv[m]  = row[470 + m * 4 + f];
    }
    float r1[25], r2[25];
    #pragma unroll
    for (int i = 0; i < 5; i++)
      #pragma unroll
      for (int j = 0; j < 5; j++) {
        float fd_ij = row[30 + (i * 5 + j) * 4 + f];
        float fd_ji = row[30 + (j * 5 + i) * 4 + f];
        r1[i * 5 + j] = ((j > i) ? fd_ij : 0.f) + ((i == j) ? d1v[i] : 0.f);
        r2[i * 5 + j] = ((j > i) ? fd_ji : 0.f) + ((i == j) ? d2v[i] : 0.f);
      }
    float qr1[75], qr2[75];
    #pragma unroll
    for (int zz = 0; zz < 15; zz++)
      #pragma unroll
      for (int m = 0; m < 5; m++) {
        float s1 = 0.f, s2 = 0.f;
        #pragma unroll
        for (int j = 0; j < 5; j++) {
          s1 += A[zz * 5 + j] * r1[j * 5 + m];
          s2 += A[zz * 5 + j] * r2[m * 5 + j];
        }
        qr1[zz * 5 + m] = s1;
        qr2[zz * 5 + m] = s2;
      }
    float hz[5];
    #pragma unroll
    for (int m = 0; m < 5; m++) {
      float s = bv[m];
      #pragma unroll
      for (int zz = 0; zz < 15; zz++) s += z[zz] * qr2[zz * 5 + m];
      hz[m] = tanhf(s);
    }
    #pragma unroll
    for (int zz = 0; zz < 15; zz++) {
      float s = z[zz];
      #pragma unroll
      for (int m = 0; m < 5; m++) s += hz[m] * qr1[zz * 5 + m];
      z[zz] = s;
    }
    #pragma unroll
    for (int m = 0; m < 5; m++) {
      float dj = (1.f - hz[m] * hz[m]) * d1v[m] * d2v[m] + 1.f;
      ldj += logf(fabsf(dj));
    }
  }
  #pragma unroll
  for (int i = 0; i < 15; i++) out[O_Z + (size_t)b * 15 + i] = z[i];
  out[O_LDJ + b] = ldj;
  #pragma unroll
  for (int i = 0; i < 32; i++)
    zh[(size_t)b * 32 + i] = (i < 15) ? (_Float16)z[i] : (_Float16)0.f;
}

// ---------------- host orchestration ----------------
extern "C" void kernel_launch(void* const* d_in, const int* in_sizes, int n_in,
                              void* d_out, int out_size, void* d_ws, size_t ws_size,
                              hipStream_t stream) {
  (void)in_sizes; (void)n_in; (void)out_size; (void)ws_size;
  P p;
  memcpy(&p, d_in, 48 * sizeof(void*));
  char* ws = (char*)d_ws;
  float* out = (float*)d_out;

  _Float16* c1w = (_Float16*)(ws + W_C1);
  _Float16* c2w = (_Float16*)(ws + W_C2);
  _Float16* c3w = (_Float16*)(ws + W_C3);
  _Float16* dw1 = (_Float16*)(ws + W_D1);
  _Float16* dw2 = (_Float16*)(ws + W_D2);
  _Float16* dw3 = (_Float16*)(ws + W_D3);
  _Float16* p1w = (_Float16*)(ws + W_P1);
  _Float16* p2w = (_Float16*)(ws + W_P2);
  _Float16* amW = (_Float16*)(ws + W_AM);
  float* amB = (float*)(ws + B_AM);
  float* p1B = (float*)(ws + B_P1);
  float* p2B = (float*)(ws + B_P2);
  float* st  = (float*)(ws + STB);

  _Float16* buf1  = (_Float16*)(ws + BUF1);   // y1/a1, later e2
  _Float16* buf2h = (_Float16*)(ws + BUF2);   // y2/a2, later e1
  float*    d2raw = (float*)(ws + BUF2);
  _Float16* hA    = (_Float16*)(ws + BUF3);   // also d2h later
  _Float16* d2h   = (_Float16*)(ws + BUF3);
  float*    amorC = (float*)(ws + OF_AMORC);
  _Float16* zh    = (_Float16*)(ws + OF_ZH);
  float*    d1raw = (float*)(ws + OF_D1RAW);
  _Float16* d1h   = (_Float16*)(ws + OF_D1H);

  prep_kernel<<<256, 256, 0, stream>>>(p, ws);

  // ---- encoder ----
  conv_gemm<12, 1, 3, 0, 150, 148, 2, 32, true, false><<<dim3(64 * 148), 256, 0, stream>>>(
      p.x, c1w, buf1, 32 * 148, nullptr, nullptr, st + S1SUM, st + S1SQ);
  finalize_bn<<<1, 256, 0, stream>>>(st + S1SUM, st + S1SQ, p.bn1_g, p.bn1_b,
                                     st + S1SC, st + S1SH, 32, 1.f / (8192.f * 148.f));
  apply_bn_act<true, 32, 148><<<4096, 256, 0, stream>>>(buf1, st + S1SC, st + S1SH, 32 * 148);
  conv_gemm<160, 32, 5, 0, 148, 144, 1, 16, false, false><<<dim3(64 * 144), 256, 0, stream>>>(
      buf1, c2w, buf2h, 16 * 144, nullptr, nullptr, st + S2SUM, st + S2SQ);
  finalize_bn<<<1, 256, 0, stream>>>(st + S2SUM, st + S2SQ, p.bn2_g, p.bn2_b,
                                     st + S2SC, st + S2SH, 16, 1.f / (8192.f * 144.f));
  apply_bn_act<true, 16, 144><<<4096, 256, 0, stream>>>(buf2h, st + S2SC, st + S2SH, 16 * 144);
  conv_gemm<112, 16, 7, 0, 144, 138, 1, 8, false, false><<<dim3(64 * 138), 256, 0, stream>>>(
      buf2h, c3w, hA, 1120, nullptr, nullptr, st + S3SUM, st + S3SQ);
  finalize_bn<<<1, 256, 0, stream>>>(st + S3SUM, st + S3SQ, p.bn3_g, p.bn3_b,
                                     st + S3SC, st + S3SH, 8, 1.f / (8192.f * 138.f));
  apply_bn_act<false, 8, 138><<<4096, 256, 0, stream>>>(hA, st + S3SC, st + S3SH, 1120);

  // ---- fused amortization heads: (8192x1120) x (1120x512) ----
  dense_gemm<4, false><<<dim3(64, 8), 256, 0, stream>>>(
      hA, 1120, amW, 512, amB, amorC, 512, 1120, nullptr, nullptr, 0);

  // ---- orthogonalize + flows ----
  flows_kernel<<<32, 256, 0, stream>>>(amorC, p.eps, out, zh);

  // ---- decoder dense ----
  dense_gemm<4, true><<<dim3(64, 2), 256, 0, stream>>>(
      zh, 32, p1w, 128, p1B, d1raw, 128, 32, st + P1SUM, st + P1SQ, 100);
  finalize_bn<<<1, 256, 0, stream>>>(st + P1SUM, st + P1SQ, p.pxbn1_g, p.pxbn1_b,
                                     st + P1SC, st + P1SH, 100, 1.f / 8192.f);
  apply_d1<<<2048, 256, 0, stream>>>(d1raw, st + P1SC, st + P1SH, d1h);
  dense_gemm<4, true><<<dim3(64, 18), 256, 0, stream>>>(
      d1h, 128, p2w, 1152, p2B, d2raw, 1152, 128, st + P2SUM, st + P2SQ, 1104);
  finalize_px<<<8, 160, 0, stream>>>(st + P2SUM, st + P2SQ, p.pxbn2_g, p.pxbn2_b,
                                     p.pxbn0_g, p.pxbn0_b, st + P2A2, st + P2C2);
  apply_d2<<<8192, 256, 0, stream>>>(d2raw, st + P2A2, st + P2C2, d2h);

  // ---- decoder transposed convs ----
  conv_gemm<56, 8, 7, 6, 138, 144, 1, 16, false, false><<<dim3(64 * 144), 256, 0, stream>>>(
      d2h, dw1, buf2h, 16 * 144, nullptr, nullptr, st + D1SUM, st + D1SQ);
  finalize_bn<<<1, 256, 0, stream>>>(st + D1SUM, st + D1SQ, p.dbn1_g, p.dbn1_b,
                                     st + D1SC, st + D1SH, 16, 1.f / (8192.f * 144.f));
  apply_bn_act<true, 16, 144><<<4096, 256, 0, stream>>>(buf2h, st + D1SC, st + D1SH, 16 * 144);
  conv_gemm<80, 16, 5, 4, 144, 148, 2, 32, false, false><<<dim3(64 * 148), 256, 0, stream>>>(
      buf2h, dw2, buf1, 32 * 148, nullptr, nullptr, st + D2SUM, st + D2SQ);
  finalize_bn<<<1, 256, 0, stream>>>(st + D2SUM, st + D2SQ, p.dbn2_g, p.dbn2_b,
                                     st + D2SC, st + D2SH, 32, 1.f / (8192.f * 148.f));
  apply_bn_act<true, 32, 148><<<4096, 256, 0, stream>>>(buf1, st + D2SC, st + D2SH, 32 * 148);
  conv_gemm<96, 32, 3, 2, 148, 150, 1, 4, false, true><<<dim3(64 * 150), 256, 0, stream>>>(
      buf1, dw3, nullptr, 0, out, p.dconv3_b, st + D2SUM, st + D2SQ);
}
